// QRNNAdapter_85581518340429
// MI455X (gfx1250) — matrix operand order repaired and checked by reading, not compiled, not run on hardware
//
#include <hip/hip_runtime.h>
#include <hip/hip_bf16.h>
#include <math.h>

// ---------------------------------------------------------------------------
// QRNN adapter for MI455X (gfx1250, wave32, WMMA).
//   HW=196, B=8, S=32, D=768, DS=192
//   Stage1: gates GEMM  M=50176 K=1536 N=384  (concat row is CONTIGUOUS:
//           [x_{t-1}, x_t] = 1536 floats starting at rowCur-768)
//   Stage2: fo-pool scan over S=32 (in-place in gates buffer)
//   Stage3: up GEMM M=50176 K=192 N=768 + bias + skip, scattered to x layout
// Memory-bound (~0.6 GB @ 23.3 TB/s vs 74 GFLOP) -> bf16 WMMA / f32 accum,
// A-strips staged once per workgroup into LDS via gfx1250 async-to-LDS copies
// and shared by 8 waves (24x / 48x A-traffic reduction vs per-wave fetch).
// ---------------------------------------------------------------------------

typedef __attribute__((ext_vector_type(16))) __bf16 bf16x16;
typedef __attribute__((ext_vector_type(8)))  float  floatx8;

#define HW_  196
#define B_   8
#define S_   32
#define D_   768
#define DS_  192
#define NJ_  (B_ * HW_)          // 1568 rows per timestep
#define M_   (S_ * NJ_)          // 50176 total rows
#define G2_  (2 * DS_)           // 384 gate cols
#define K1_  (2 * D_)            // 1536
#define NT1_ (G2_ / 16)          // 24 nTiles, 3 per wave
#define MT_  (M_ / 16)           // 3136 mTiles
#define KT2_ (DS_ / 32)          // 6
#define NT2_ (D_ / 16)           // 48 nTiles, 6 per wave

// flat index of xr[s, j, 0] inside x:  j = b*HW + hw ; flat = (hw*B*S + b*S + s)*D
__device__ __forceinline__ size_t xr_row_off(int s, int j) {
    int hw = j % HW_;
    int b  = j / HW_;
    return ((size_t)hw * (B_ * S_) + (size_t)b * S_ + (size_t)s) * D_;
}

// gfx1250 async copy: 16 bytes global -> LDS, tracked by ASYNCcnt (per wave).
__device__ __forceinline__ void async_ld_b128(uint32_t lds_byte_addr,
                                              const void* gaddr) {
    asm volatile("global_load_async_to_lds_b128 %0, %1, off"
                 :: "v"(lds_byte_addr), "v"(gaddr)
                 : "memory");
}
__device__ __forceinline__ void wait_async0() {
    asm volatile("s_wait_asynccnt 0" ::: "memory");
}
// low 32 bits of a generic LDS pointer == wave-relative LDS byte address
__device__ __forceinline__ uint32_t lds_addr(const void* p) {
    return (uint32_t)(uintptr_t)p;
}

// ---------------------------------------------------------------------------
// Pack fp32 row-major (K x N) weight into bf16 B-operand register layout:
//   pack[(((kt*numN)+nt)*32 + l)*16 + e] = W[kt*32 + (l/16)*8 + e + (e&8)][nt*16 + l%16]
// ---------------------------------------------------------------------------
__global__ void qrnn_pack_w(const float* __restrict__ W, __bf16* __restrict__ pack,
                            int K, int N) {
    int numN  = N / 16;
    int total = (K / 32) * numN * 32;
    int id = blockIdx.x * blockDim.x + threadIdx.x;
    if (id >= total) return;
    int lane = id & 31;
    int tile = id >> 5;
    int nt = tile % numN;
    int kt = tile / numN;
    const float* src = W + (size_t)(kt * 32 + (lane >> 4) * 8) * N
                         + (size_t)nt * 16 + (lane & 15);
    __bf16* dst = pack + (size_t)id * 16;
#pragma unroll
    for (int e = 0; e < 16; ++e)
        dst[e] = (__bf16)src[(size_t)(e + (e & 8)) * N];
}

// ---------------------------------------------------------------------------
// Stage 1: gates = [x_prev|x_cur] @ W_zf + b_zf.
// One block (8 waves) per mTile; A strip (16 x 1536 f32, 96 KB) staged into
// LDS with async copies; wave w computes nTiles 3w..3w+2.
// ---------------------------------------------------------------------------
extern __shared__ float smem[];

__global__ __launch_bounds__(256)
void qrnn_gates_gemm(const float* __restrict__ x,
                     const __bf16* __restrict__ wpack,
                     const float* __restrict__ b_zf,
                     float* __restrict__ gates) {
    const int t     = threadIdx.x;
    const int row0  = blockIdx.x * 16;
    const int s     = row0 / NJ_;          // uniform over block (1568 % 16 == 0)
    const int jBase = row0 % NJ_;

    // ---- async-stage A strip: row r = 1536 floats at x[row(s,jBase+r)] - 768.
    // s==0: only k in [768,1536) is valid (the x_cur half); kt<24 is skipped.
    {
        const int r = t >> 4;              // 0..15
        const int c = t & 15;
        const int kLo = (s == 0) ? D_ : 0;
        const float* grow = x + xr_row_off(s, jBase + r) - D_;
        const uint32_t lrow = lds_addr(smem) + (uint32_t)r * (K1_ * 4);
        for (int k = kLo + c * 4; k < K1_; k += 64)       // 16 thr x 4 floats
            async_ld_b128(lrow + (uint32_t)k * 4, grow + k);
    }
    wait_async0();
    __syncthreads();

    const int wave = t >> 5;
    const int lane = t & 31;
    const int m    = lane & 15;
    const int half = lane >> 4;
    const int n0   = wave * 3;

    const float* aRow = smem + (size_t)m * K1_;
    floatx8 acc[3];
#pragma unroll
    for (int q = 0; q < 3; ++q)
        acc[q] = (floatx8){0.f, 0.f, 0.f, 0.f, 0.f, 0.f, 0.f, 0.f};

    for (int kt = (s == 0 ? 24 : 0); kt < K1_ / 32; ++kt) {
        const float* p0 = aRow + kt * 32 + half * 8;      // K = half*8 + 0..7
        bf16x16 a;
#pragma unroll
        for (int e = 0; e < 8; ++e) {
            a[e]     = (__bf16)p0[e];
            a[8 + e] = (__bf16)p0[16 + e];                // K = 16 + half*8 + e
        }
        const __bf16* bp = wpack + ((size_t)(kt * NT1_ + n0) * 32 + lane) * 16;
#pragma unroll
        for (int q = 0; q < 3; ++q) {
            const bf16x16 bm = *(const bf16x16*)(bp + (size_t)q * 32 * 16);
            acc[q] = __builtin_amdgcn_wmma_f32_16x16x32_bf16(
                false, a, false, bm, (short)0, acc[q], false, false);
        }
    }

#pragma unroll
    for (int q = 0; q < 3; ++q) {
        int col = (n0 + q) * 16 + m;                      // N = lane%16
        float bias = b_zf[col];
        float* grow = gates + ((size_t)row0 + half * 8) * G2_ + col;
#pragma unroll
        for (int r = 0; r < 8; ++r)                       // M = half*8 + r
            grow[(size_t)r * G2_] = acc[q][r] + bias;
    }
}

// ---------------------------------------------------------------------------
// Stage 2: fo-pool. h_t = sig(f)*h_{t-1} + (1-sig(f))*tanh(z); overwrites the
// z-slot of gates with h. One thread per (j, c).
// ---------------------------------------------------------------------------
__global__ void qrnn_scan(float* __restrict__ gates) {
    int id = blockIdx.x * blockDim.x + threadIdx.x;
    if (id >= NJ_ * DS_) return;
    float* p = gates + (size_t)(id / DS_) * G2_ + (id % DS_);
    const size_t step = (size_t)NJ_ * G2_;
    float h = 0.f;
#pragma unroll 4
    for (int s = 0; s < S_; ++s) {
        float z  = p[0];
        float f  = p[DS_];
        float ft = 1.f / (1.f + __expf(-f));
        h = ft * h + (1.f - ft) * tanhf(z);
        p[0] = h;
        p += step;
    }
}

// ---------------------------------------------------------------------------
// Stage 3: out = xr + h @ W_up + b_up in original x layout.
// One block per mTile; A strip (16 x 192 f32 from gates rows, stride 384)
// async-staged to LDS; wave w computes nTiles 6w..6w+5.
// ---------------------------------------------------------------------------
__global__ __launch_bounds__(256)
void qrnn_out_gemm(const float* __restrict__ x,
                   const float* __restrict__ gates,
                   const __bf16* __restrict__ wpack,
                   const float* __restrict__ b_up,
                   float* __restrict__ out) {
    const int t     = threadIdx.x;
    const int row0  = blockIdx.x * 16;
    const int s     = row0 / NJ_;
    const int jBase = row0 % NJ_;

    {   // async-stage 16 x 192 f32 (h lives in cols 0..191, row stride 384)
        const int r = t >> 4;
        const int c = t & 15;
        const float* grow = gates + (size_t)(row0 + r) * G2_;
        const uint32_t lrow = lds_addr(smem) + (uint32_t)r * (DS_ * 4);
        for (int k = c * 4; k < DS_; k += 64)             // 3 iterations
            async_ld_b128(lrow + (uint32_t)k * 4, grow + k);
    }
    wait_async0();
    __syncthreads();

    const int wave = t >> 5;
    const int lane = t & 31;
    const int m    = lane & 15;
    const int half = lane >> 4;
    const int n0   = wave * 6;

    const float* aRow = smem + (size_t)m * DS_;
    floatx8 acc[6];
#pragma unroll
    for (int q = 0; q < 6; ++q)
        acc[q] = (floatx8){0.f, 0.f, 0.f, 0.f, 0.f, 0.f, 0.f, 0.f};

#pragma unroll
    for (int kt = 0; kt < KT2_; ++kt) {
        const float* p0 = aRow + kt * 32 + half * 8;
        bf16x16 a;
#pragma unroll
        for (int e = 0; e < 8; ++e) {
            a[e]     = (__bf16)p0[e];
            a[8 + e] = (__bf16)p0[16 + e];
        }
        const __bf16* bp = wpack + ((size_t)(kt * NT2_ + n0) * 32 + lane) * 16;
#pragma unroll
        for (int q = 0; q < 6; ++q) {
            const bf16x16 bm = *(const bf16x16*)(bp + (size_t)q * 32 * 16);
            acc[q] = __builtin_amdgcn_wmma_f32_16x16x32_bf16(
                false, a, false, bm, (short)0, acc[q], false, false);
        }
    }

#pragma unroll
    for (int q = 0; q < 6; ++q) {
        int col = (n0 + q) * 16 + m;
        float bias = b_up[col];
#pragma unroll
        for (int r = 0; r < 8; ++r) {
            int j_r = jBase + half * 8 + r;               // M = half*8 + r
            size_t flat = xr_row_off(s, j_r) + col;
            out[flat] = x[flat] + acc[q][r] + bias;
        }
    }
}

// ---------------------------------------------------------------------------
// Host launcher. Workspace (bytes):
//   [0, 1179648)           W_zf packed bf16 (1536*384)
//   [1179648, 1474560)     W_up packed bf16 (192*768)
//   [1474560, +77.07 MB)   gates fp32 (50176 * 384)
// ---------------------------------------------------------------------------
extern "C" void kernel_launch(void* const* d_in, const int* in_sizes, int n_in,
                              void* d_out, int out_size, void* d_ws, size_t ws_size,
                              hipStream_t stream) {
    const float* x    = (const float*)d_in[0];
    const float* W_zf = (const float*)d_in[1];
    const float* b_zf = (const float*)d_in[2];
    const float* W_up = (const float*)d_in[3];
    const float* b_up = (const float*)d_in[4];
    float* out = (float*)d_out;

    __bf16* wzfPack = (__bf16*)d_ws;
    __bf16* wupPack = wzfPack + (size_t)K1_ * G2_;        // +589824 elems
    float*  gates   = (float*)((char*)d_ws + 1474560);

    {   // weight packing (tiny, once per launch)
        int total1 = (K1_ / 32) * NT1_ * 32;              // 36864
        qrnn_pack_w<<<(total1 + 255) / 256, 256, 0, stream>>>(W_zf, wzfPack, K1_, G2_);
        int total2 = KT2_ * NT2_ * 32;                    // 9216
        qrnn_pack_w<<<(total2 + 255) / 256, 256, 0, stream>>>(W_up, wupPack, DS_, D_);
    }

    // stage 1: one block per mTile, A strip in LDS (16*1536*4 = 96 KB)
    qrnn_gates_gemm<<<MT_, 256, 16 * K1_ * sizeof(float), stream>>>(
        x, wzfPack, b_zf, gates);

    // stage 2: recurrent fo-pool
    qrnn_scan<<<(NJ_ * DS_ + 255) / 256, 256, 0, stream>>>(gates);

    // stage 3: one block per mTile, h strip in LDS (16*192*4 = 12 KB)
    qrnn_out_gemm<<<MT_, 256, 16 * DS_ * sizeof(float), stream>>>(
        x, gates, wupPack, b_up, out);
}
